// TestSiluMul_48077863912199
// MI455X (gfx1250) — 8-bit matrix operand order repaired and checked by reading, compiled, not run on hardware
//
#include <hip/hip_runtime.h>
#include <hip/hip_bf16.h>

typedef __attribute__((ext_vector_type(16))) int   v16i;
typedef __attribute__((ext_vector_type(8)))  float v8f;
typedef __attribute__((ext_vector_type(4)))  int   v4i;
typedef __attribute__((ext_vector_type(2)))  int   v2i;

#define HID    4096
#define XROW   (2*HID)
#define LPITCH 144
#define BM     128
#define BN     128
#define BK     128
#define NK     (HID / BK)

#define WAIT_ASYNC(n) asm volatile("s_wait_asynccnt %0" :: "n"(n) : "memory")

__device__ __forceinline__ unsigned int f32_to_e4m3(float f) {
    unsigned int s = (__float_as_uint(f) >> 24) & 0x80u;
    float a = fabsf(f);
    a = fminf(a, 448.0f);
    if (a < 0.015625f) {
        int mi = (int)rintf(a * 512.0f);
        return s | (unsigned int)mi;
    }
    unsigned int bits = __float_as_uint(a);
    int e = (int)(bits >> 23) - 127;
    unsigned int mant = (bits >> 20) & 0x7u;
    unsigned int rem  = bits & 0xFFFFFu;
    mant += (rem > 0x80000u || (rem == 0x80000u && (mant & 1u))) ? 1u : 0u;
    unsigned int enc = ((unsigned int)(e + 7) << 3) + mant;
    return s | enc;
}

__global__ __launch_bounds__(256) void silu_mul_quant_kernel(
    const float* __restrict__ x, const float* __restrict__ xs,
    unsigned int* __restrict__ yq, long long total  ) {
    long long idx = (long long)blockIdx.x * blockDim.x + threadIdx.x;
    if (idx >= total) return;
    long long t = idx / (HID / 4);
    int h4      = (int)(idx % (HID / 4));
    const float4 g = *(const float4*)(x + t * XROW + h4 * 4);
    const float4 u = *(const float4*)(x + t * XROW + HID + h4 * 4);
    float inv = 1.0f / xs[0];
    float y0 = (g.x / (1.0f + expf(-g.x))) * u.x * inv;
    float y1 = (g.y / (1.0f + expf(-g.y))) * u.y * inv;
    float y2 = (g.z / (1.0f + expf(-g.z))) * u.z * inv;
    float y3 = (g.w / (1.0f + expf(-g.w))) * u.w * inv;
    unsigned int pk = f32_to_e4m3(y0) | (f32_to_e4m3(y1) << 8) |
                      (f32_to_e4m3(y2) << 16) | (f32_to_e4m3(y3) << 24);
    yq[idx] = pk;
}

__global__ __launch_bounds__(256) void w_quant_kernel(
    const float* __restrict__ w, const float* __restrict__ ws,
    unsigned int* __restrict__ wq, long long total  ) {
    long long idx = (long long)blockIdx.x * blockDim.x + threadIdx.x;
    if (idx >= total) return;
    const float4 v = *(const float4*)(w + idx * 4);
    float inv = 1.0f / ws[0];
    unsigned int pk = f32_to_e4m3(v.x * inv) | (f32_to_e4m3(v.y * inv) << 8) |
                      (f32_to_e4m3(v.z * inv) << 16) | (f32_to_e4m3(v.w * inv) << 24);
    wq[idx] = pk;
}

__device__ __forceinline__ void async_cp16(const unsigned char* g, const unsigned char* l) {
    unsigned lds_off = (unsigned)(unsigned long long)(const void*)l;
    unsigned long long ga = (unsigned long long)g;
    asm volatile("global_load_async_to_lds_b128 %0, %1, off" :: "v"(lds_off), "v"(ga) : "memory");
}

__global__ __launch_bounds__(256) void gemm_fp8_wmma_kernel(
    const unsigned char* __restrict__ Aq,
    const unsigned char* __restrict__ Bq,
    float* __restrict__ C,
    const float* __restrict__ xs, const float* __restrict__ ws) {
    __shared__ __align__(16) unsigned char lds[2][(BM + BN) * LPITCH];

    const int tid   = threadIdx.x;
    const int lane  = tid & 31;
    const int wave  = tid >> 5;
    const int m_w   = (wave >> 1) * 32;
    const int n_w   = (wave & 1) * 64;
    const int lrow  = lane & 15;
    const int lsel  = lane >> 4;

    const int m0 = blockIdx.y * BM;
    const int n0 = blockIdx.x * BN;

    const int crow = tid >> 3;
    const int ccol = (tid & 7) * 16;

    const unsigned char* gA = Aq + (size_t)(m0 + crow) * HID + ccol;
    const unsigned char* gB = Bq + (size_t)(n0 + crow) * HID + ccol;

    {
        unsigned char* la = &lds[0][crow * LPITCH + ccol];
        unsigned char* lb = &lds[0][(BM + crow) * LPITCH + ccol];
        #pragma unroll
        for (int s = 0; s < 4; ++s) {
            async_cp16(gA + (size_t)(s * 32) * HID, la + s * 32 * LPITCH);
            async_cp16(gB + (size_t)(s * 32) * HID, lb + s * 32 * LPITCH);
        }
    }

    v8f acc[2][4];
    #pragma unroll
    for (int mt = 0; mt < 2; ++mt)
        #pragma unroll
        for (int nt = 0; nt < 4; ++nt)
            acc[mt][nt] = (v8f){};

    for (int kk = 0; kk < NK; ++kk) {
        const int cur = kk & 1;
        if (kk + 1 < NK) {
            const int koff = (kk + 1) * BK;
            unsigned char* la = &lds[cur ^ 1][crow * LPITCH + ccol];
            unsigned char* lb = &lds[cur ^ 1][(BM + crow) * LPITCH + ccol];
            #pragma unroll
            for (int s = 0; s < 4; ++s) {
                async_cp16(gA + koff + (size_t)(s * 32) * HID, la + s * 32 * LPITCH);
                async_cp16(gB + koff + (size_t)(s * 32) * HID, lb + s * 32 * LPITCH);
            }
            WAIT_ASYNC(8);
        } else {
            WAIT_ASYNC(0);
        }
        __syncthreads();

        const unsigned char* bufA = &lds[cur][0];
        const unsigned char* bufB = &lds[cur][BM * LPITCH];

        v16i bfrag[4];
        #pragma unroll
        for (int nt = 0; nt < 4; ++nt) {
            const unsigned char* p = bufB + (n_w + nt * 16 + lrow) * LPITCH + 16 * lsel;
            #pragma unroll
            for (int j = 0; j < 4; ++j) {
                v4i t = *(const v4i*)(p + 32 * j);
                bfrag[nt][4 * j]     = t[0];
                bfrag[nt][4 * j + 1] = t[1];
                bfrag[nt][4 * j + 2] = t[2];
                bfrag[nt][4 * j + 3] = t[3];
            }
        }
        #pragma unroll
        for (int mt = 0; mt < 2; ++mt) {
            v16i afrag;
            const unsigned char* pa = bufA + (m_w + mt * 16 + lrow) * LPITCH + 16 * lsel;
            #pragma unroll
            for (int i = 0; i < 8; ++i) {
                v2i t = *(const v2i*)(pa + 32 * (i >> 1) + 8 * (i & 1));
                afrag[2 * i]     = t[0];
                afrag[2 * i + 1] = t[1];
            }
            #pragma unroll
            for (int nt = 0; nt < 4; ++nt)
                acc[mt][nt] = __builtin_amdgcn_wmma_f32_16x16x128_fp8_fp8(
                    afrag, bfrag[nt], (short)0, acc[mt][nt], false, false);
        }
        __syncthreads();
    }

    const float scale = xs[0] * ws[0];
    #pragma unroll
    for (int mt = 0; mt < 2; ++mt) {
        float* Crow = C + (size_t)(m0 + m_w + mt * 16 + 8 * lsel) * HID + n0 + n_w + lrow;
        #pragma unroll
        for (int nt = 0; nt < 4; ++nt) {
            #pragma unroll
            for (int r = 0; r < 8; ++r) {
                Crow[(size_t)r * HID + nt * 16] = acc[mt][nt][r] * scale;
            }
        }
    }
}

extern "C" void kernel_launch(void* const* d_in, const int* in_sizes, int n_in,
                              void* d_out, int out_size, void* d_ws, size_t ws_size,
                              hipStream_t stream) {
    const float* x  = (const float*)d_in[0];
    const float* w  = (const float*)d_in[1];
    const float* xs = (const float*)d_in[2];
    const float* ws = (const float*)d_in[3];
    float* out = (float*)d_out;

    const long long T = (long long)in_sizes[0] / XROW;

    unsigned char* yq = (unsigned char*)d_ws;
    unsigned char* wq = yq + (size_t)T * HID;

    {
        long long total = T * (HID / 4);
        int grid = (int)((total + 255) / 256);
        silu_mul_quant_kernel<<<grid, 256, 0, stream>>>(x, xs, (unsigned int*)yq, total);
    }
    {
        long long total = (long long)HID * HID / 4;
        int grid = (int)((total + 255) / 256);
        w_quant_kernel<<<grid, 256, 0, stream>>>(w, ws, (unsigned int*)wq, total);
    }
    {
        dim3 grid(HID / BN, (unsigned)(T / BM), 1);
        gemm_fp8_wmma_kernel<<<grid, 256, 0, stream>>>(yq, wq, out, xs, ws);
    }
}
